// QAModel_71365176590350
// MI455X (gfx1250) — matrix operand order repaired and checked by reading, not compiled, not run on hardware
//
#include <hip/hip_runtime.h>
#include <hip/hip_bf16.h>

// ---------------------------------------------------------------------------
// BiDAF QA model forward pass for MI455X (gfx1250, wave32, WMMA).
//
// Shapes: V=50000 E=100 H=128 H2=256 B=64 CL=512 QL=64, 4H=512 gate width.
// Heavy GEMMs (x@Wih^T) and the LSTM recurrence (h@Whh^T) run on
// v_wmma_f32_16x16x32_f16 (f16 inputs, f32 accumulate).
//  - GEMM: each wave computes a 16x64 output strip: one A fragment reused
//    across 4 B tiles (4 accumulators, 4 WMMAs per K-step).
//  - Recurrence: persistent 1024-thread workgroup per direction, and both
//    directions of a layer run CONCURRENTLY (grid=2, blockIdx = direction,
//    each with its own xW stream / Whh / reversed time order). Whh (128 KB
//    f16) staged into LDS once; h(f16)/c(f32)/gates(f32) LDS-resident.
//    Total 304 KB of the WGP's 320 KB. Per step: 128 tiles = 16 WMMAs/wave,
//    all matrix operands from LDS.
// ---------------------------------------------------------------------------

typedef __attribute__((ext_vector_type(16))) _Float16 v16h;
typedef __attribute__((ext_vector_type(8)))  float    v8f;

#define NEGV (-1000000000.0f)

__device__ __forceinline__ float sigf(float x) { return 1.0f / (1.0f + __expf(-x)); }

// ---------------- f32 -> f16 convert with column zero-pad --------------------
__global__ void k_pad16(const float* __restrict__ src, _Float16* __restrict__ dst,
                        int R, int C, int Cp) {
  int i = blockIdx.x * blockDim.x + threadIdx.x;
  if (i >= R * Cp) return;
  int r = i / Cp, c = i - r * Cp;
  dst[i] = (_Float16)((c < C) ? src[(size_t)r * C + c] : 0.0f);
}

// ---------------- embedding gather -> padded f16 -----------------------------
__global__ void k_embed(const float* __restrict__ emb, const int* __restrict__ idx,
                        _Float16* __restrict__ dst, int rows, int E, int Kp) {
  int i = blockIdx.x * blockDim.x + threadIdx.x;
  if (i >= rows * Kp) return;
  int r = i / Kp, k = i - r * Kp;
  float v = (k < E) ? emb[(size_t)idx[r] * E + k] : 0.0f;
  dst[i] = (_Float16)v;
}

// ---------------- WMMA GEMM: C[M,N] = X[M,Kp] * W[N,Kp]^T + bias -------------
// X, W are f16 row-major. Each wave computes a 16x64 strip (4 tiles, A reuse).
// A/B fragments use contiguous k = 16*(lane/16) + j (same for both); f32 C/D per CDNA5 ISA 7.12.2.
__global__ __launch_bounds__(256) void k_gemm_f16(
    const _Float16* __restrict__ X, const _Float16* __restrict__ W,
    const float* __restrict__ bias, float* __restrict__ C,
    int M, int Kp, int N) {
  const int lane = threadIdx.x & 31;
  const int wave = threadIdx.x >> 5;
  const int nstrips = N >> 6;                    // 64-wide strips
  const int total = (M >> 4) * nstrips;
  int tid = blockIdx.x * 8 + wave;
  if (tid >= total) return;                      // wave-uniform: EXEC all-ones
  int mt = tid / nstrips, ns = tid - mt * nstrips;
  int ntbase = ns * 4;
  const _Float16* Xr = X + (size_t)(mt * 16 + (lane & 15)) * Kp;
  const _Float16* Wr = W + (size_t)(lane & 15) * Kp;   // + (nt*16)*Kp per tile
  v8f acc0 = {}, acc1 = {}, acc2 = {}, acc3 = {};
  for (int k0 = 0; k0 < Kp; k0 += 32) {
    v16h a;
#pragma unroll
    for (int j = 0; j < 16; ++j) {
      int v = j >> 1;
      int ka = ((lane & 16) ? 16 : 0) + (v << 1) + (j & 1);
      a[j] = Xr[k0 + ka];
    }
    v16h bf0, bf1, bf2, bf3;
#pragma unroll
    for (int j = 0; j < 16; ++j) {
      int v = j >> 1;
      int kb = ((lane & 16) ? 16 : 0) + (v << 1) + (j & 1);
      bf0[j] = Wr[(size_t)(ntbase + 0) * 16 * Kp + k0 + kb];
      bf1[j] = Wr[(size_t)(ntbase + 1) * 16 * Kp + k0 + kb];
      bf2[j] = Wr[(size_t)(ntbase + 2) * 16 * Kp + k0 + kb];
      bf3[j] = Wr[(size_t)(ntbase + 3) * 16 * Kp + k0 + kb];
    }
    acc0 = __builtin_amdgcn_wmma_f32_16x16x32_f16(false, a, false, bf0, (short)0, acc0, false, false);
    acc1 = __builtin_amdgcn_wmma_f32_16x16x32_f16(false, a, false, bf1, (short)0, acc1, false, false);
    acc2 = __builtin_amdgcn_wmma_f32_16x16x32_f16(false, a, false, bf2, (short)0, acc2, false, false);
    acc3 = __builtin_amdgcn_wmma_f32_16x16x32_f16(false, a, false, bf3, (short)0, acc3, false, false);
  }
  v8f* accs[4] = {&acc0, &acc1, &acc2, &acc3};
#pragma unroll
  for (int u = 0; u < 4; ++u) {
    int n = (ntbase + u) * 16 + (lane & 15);
    float bv = bias ? bias[n] : 0.0f;
    v8f acc = *accs[u];
#pragma unroll
    for (int r = 0; r < 8; ++r) {
      int m = mt * 16 + r + ((lane & 16) ? 8 : 0);
      C[(size_t)m * N + n] = acc[r] + bv;
    }
  }
}

// ---------------- persistent LSTM recurrence (both directions) ---------------
// grid = 2 blocks; blockIdx.x = direction (0=fwd, 1=bwd), each on its own WGP
// with its own xW stream and Whh. xW: (B*T, 512) f32, row = b*T + t.
// out: (B,T,2H) f32, direction selects the 128-wide half.
__global__ __launch_bounds__(1024) void k_lstm2(
    const float* __restrict__ xW0, const float* __restrict__ xW1,
    const _Float16* __restrict__ Whh0, const _Float16* __restrict__ Whh1,
    float* __restrict__ out, int T) {
  const int dir = blockIdx.x;
  const float* __restrict__ xW = dir ? xW1 : xW0;
  const _Float16* __restrict__ Whh = dir ? Whh1 : Whh0;
  const int reverse = dir;
  extern __shared__ char smem[];
  _Float16* hbuf = (_Float16*)smem;                                   // 64*128 f16  (16 KB)
  float* cbuf = (float*)(smem + 64 * 128 * sizeof(_Float16));         // 64*128 f32  (32 KB)
  float* gbuf = cbuf + 64 * 128;                                      // 64*512 f32  (128 KB)
  _Float16* whhl = (_Float16*)(gbuf + 64 * 512);                      // 512*128 f16 (128 KB)
  const int lane = threadIdx.x & 31;
  const int wave = threadIdx.x >> 5;
  for (int idx = threadIdx.x; idx < 64 * 128; idx += 1024) {
    hbuf[idx] = (_Float16)0.0f;
    cbuf[idx] = 0.0f;
  }
  // stage Whh into LDS (dword copies)
  for (int idx = threadIdx.x; idx < 512 * 128 / 2; idx += 1024)
    ((unsigned int*)whhl)[idx] = ((const unsigned int*)Whh)[idx];
  __syncthreads();
  const int mt = wave >> 3;              // 0..3  (M block of 16 rows)
  const int ntbase = (wave & 7) * 4;     // 4 consecutive N tiles per wave
  const int mrow = mt * 16 + (lane & 15);
  for (int t = 0; t < T; ++t) {
    int tt = reverse ? (T - 1 - t) : t;
    v8f acc0 = {}, acc1 = {}, acc2 = {}, acc3 = {};
#pragma unroll
    for (int k0 = 0; k0 < 128; k0 += 32) {
      v16h a;
#pragma unroll
      for (int j = 0; j < 16; ++j) {
        int v = j >> 1;
        int ka = ((lane & 16) ? 16 : 0) + (v << 1) + (j & 1);
        a[j] = hbuf[mrow * 128 + k0 + ka];
      }
      v16h bf0, bf1, bf2, bf3;
#pragma unroll
      for (int j = 0; j < 16; ++j) {
        int v = j >> 1;
        int kb = ((lane & 16) ? 16 : 0) + (v << 1) + (j & 1);
        int col = (lane & 15) * 128;
        bf0[j] = whhl[((ntbase + 0) * 16) * 128 + col + k0 + kb];
        bf1[j] = whhl[((ntbase + 1) * 16) * 128 + col + k0 + kb];
        bf2[j] = whhl[((ntbase + 2) * 16) * 128 + col + k0 + kb];
        bf3[j] = whhl[((ntbase + 3) * 16) * 128 + col + k0 + kb];
      }
      acc0 = __builtin_amdgcn_wmma_f32_16x16x32_f16(false, a, false, bf0, (short)0, acc0, false, false);
      acc1 = __builtin_amdgcn_wmma_f32_16x16x32_f16(false, a, false, bf1, (short)0, acc1, false, false);
      acc2 = __builtin_amdgcn_wmma_f32_16x16x32_f16(false, a, false, bf2, (short)0, acc2, false, false);
      acc3 = __builtin_amdgcn_wmma_f32_16x16x32_f16(false, a, false, bf3, (short)0, acc3, false, false);
    }
    {
      v8f* accs[4] = {&acc0, &acc1, &acc2, &acc3};
#pragma unroll
      for (int u = 0; u < 4; ++u) {
        int n = (ntbase + u) * 16 + (lane & 15);
        v8f acc = *accs[u];
#pragma unroll
        for (int r = 0; r < 8; ++r) {
          int m = mt * 16 + r + ((lane & 16) ? 8 : 0);
          gbuf[m * 512 + n] = acc[r] + xW[((size_t)m * T + tt) * 512 + n];
        }
      }
    }
    __syncthreads();
    for (int idx = threadIdx.x; idx < 64 * 128; idx += 1024) {
      int b = idx >> 7, hh = idx & 127;
      float ig = gbuf[b * 512 + hh];
      float fg = gbuf[b * 512 + 128 + hh];
      float gg = gbuf[b * 512 + 256 + hh];
      float og = gbuf[b * 512 + 384 + hh];
      float c = sigf(fg) * cbuf[idx] + sigf(ig) * tanhf(gg);
      float h = sigf(og) * tanhf(c);
      cbuf[idx] = c;
      hbuf[idx] = (_Float16)h;
      out[((size_t)b * T + tt) * 256 + dir * 128 + hh] = h;
    }
    __syncthreads();
  }
}

// ---------------- BiDAF attention; writes g as f16 (mod1 GEMM input) ---------
__global__ __launch_bounds__(256) void k_bidaf(
    const float* __restrict__ c_enc, const float* __restrict__ q_enc,
    const int* __restrict__ cmask, const int* __restrict__ qmask,
    const float* __restrict__ att_w, float* __restrict__ c2qs,
    _Float16* __restrict__ gout) {
  extern __shared__ char smem[];
  float* qbuf  = (float*)smem;        // 64*256
  float* wc    = qbuf + 64 * 256;     // 256
  float* wq    = wc + 256;            // 256
  float* wm    = wq + 256;            // 256
  float* qw    = wm + 256;            // 64
  float* rowmx = qw + 64;             // 512
  float* cmrow = rowmx + 512;         // 8*256
  float* arow  = cmrow + 8 * 256;     // 8*64
  float* red   = arow + 8 * 64;       // 256
  float* q2c   = red + 256;           // 256
  const int b = blockIdx.x;
  const int lane = threadIdx.x & 31;
  const int wave = threadIdx.x >> 5;
  for (int i = threadIdx.x; i < 64 * 256; i += 256) qbuf[i] = q_enc[(size_t)b * 64 * 256 + i];
  if (threadIdx.x < 256) {
    wc[threadIdx.x] = att_w[threadIdx.x];
    wq[threadIdx.x] = att_w[256 + threadIdx.x];
    wm[threadIdx.x] = att_w[512 + threadIdx.x];
  }
  __syncthreads();
  if (threadIdx.x < 64) {
    float s = 0.f;
    for (int h = 0; h < 256; ++h) s += qbuf[threadIdx.x * 256 + h] * wq[h];
    qw[threadIdx.x] = s;
  }
  __syncthreads();
  for (int crow = wave; crow < 512; crow += 8) {
    const float* crp = c_enc + ((size_t)b * 512 + crow) * 256;
    float cw = 0.f;
#pragma unroll
    for (int u = 0; u < 8; ++u) {
      float v = crp[u * 32 + lane];
      cmrow[wave * 256 + u * 32 + lane] = v * wm[u * 32 + lane];
      cw += v * wc[u * 32 + lane];
    }
#pragma unroll
    for (int o = 16; o > 0; o >>= 1) cw += __shfl_xor(cw, o, 32);
    float s0 = cw + qw[lane], s1 = cw + qw[lane + 32];
    float d0 = 0.f, d1 = 0.f;
    for (int h = 0; h < 256; ++h) {
      float cm = cmrow[wave * 256 + h];
      d0 += cm * qbuf[lane * 256 + h];
      d1 += cm * qbuf[(lane + 32) * 256 + h];
    }
    s0 += d0; s1 += d1;
    if (qmask[b * 64 + lane] == 0) s0 = NEGV;
    if (qmask[b * 64 + lane + 32] == 0) s1 = NEGV;
    float mx = fmaxf(s0, s1);
#pragma unroll
    for (int o = 16; o > 0; o >>= 1) mx = fmaxf(mx, __shfl_xor(mx, o, 32));
    float e0 = __expf(s0 - mx), e1 = __expf(s1 - mx);
    float sum = e0 + e1;
#pragma unroll
    for (int o = 16; o > 0; o >>= 1) sum += __shfl_xor(sum, o, 32);
    arow[wave * 64 + lane] = e0 / sum;
    arow[wave * 64 + lane + 32] = e1 / sum;
    if (lane == 0) rowmx[crow] = (cmask[b * 512 + crow] == 0) ? NEGV : mx;
#pragma unroll
    for (int u = 0; u < 8; ++u) {
      int h = u * 32 + lane;
      float acc = 0.f;
      for (int q = 0; q < 64; ++q) acc += arow[wave * 64 + q] * qbuf[q * 256 + h];
      c2qs[((size_t)b * 512 + crow) * 256 + h] = acc;
    }
  }
  __syncthreads();
  // b_att softmax over c
  red[threadIdx.x] = fmaxf(rowmx[threadIdx.x], rowmx[threadIdx.x + 256]);
  __syncthreads();
  for (int s = 128; s > 0; s >>= 1) {
    if (threadIdx.x < s) red[threadIdx.x] = fmaxf(red[threadIdx.x], red[threadIdx.x + s]);
    __syncthreads();
  }
  float gmax = red[0];
  __syncthreads();
  red[threadIdx.x] = __expf(rowmx[threadIdx.x] - gmax) + __expf(rowmx[threadIdx.x + 256] - gmax);
  __syncthreads();
  for (int s = 128; s > 0; s >>= 1) {
    if (threadIdx.x < s) red[threadIdx.x] += red[threadIdx.x + s];
    __syncthreads();
  }
  float gden = red[0];
  __syncthreads();
  {
    int h = threadIdx.x;
    float acc = 0.f;
    for (int c = 0; c < 512; ++c)
      acc += (__expf(rowmx[c] - gmax) / gden) * c_enc[((size_t)b * 512 + c) * 256 + h];
    q2c[h] = acc;
  }
  __syncthreads();
  for (int idx = threadIdx.x; idx < 512 * 1024; idx += 256) {
    int crow = idx >> 10, j = idx & 1023;
    int h = j & 255, part = j >> 8;
    float cv = c_enc[((size_t)b * 512 + crow) * 256 + h];
    float c2 = c2qs[((size_t)b * 512 + crow) * 256 + h];
    float val = (part == 0) ? cv : (part == 1) ? c2 : (part == 2) ? cv * c2 : cv * q2c[h];
    gout[((size_t)b * 512 + crow) * 1024 + j] = (_Float16)val;
  }
}

// ---------------- final masked projections -----------------------------------
__global__ void k_project(const float* __restrict__ m2, const float* __restrict__ sw,
                          const float* __restrict__ sb, const float* __restrict__ ew,
                          const float* __restrict__ eb, const int* __restrict__ cmask,
                          float* __restrict__ outp) {
  int i = blockIdx.x * blockDim.x + threadIdx.x;
  if (i >= 64 * 512) return;
  const float* row = m2 + (size_t)i * 256;
  float s = 0.f, e = 0.f;
  for (int h = 0; h < 256; ++h) { float v = row[h]; s += v * sw[h]; e += v * ew[h]; }
  s += sb[0]; e += eb[0];
  if (cmask[i] == 0) { s = NEGV; e = NEGV; }
  outp[i] = s;
  outp[64 * 512 + i] = e;
}

// ---------------------------------------------------------------------------
extern "C" void kernel_launch(void* const* d_in, const int* in_sizes, int n_in,
                              void* d_out, int out_size, void* d_ws, size_t ws_size,
                              hipStream_t stream) {
  (void)in_sizes; (void)n_in; (void)out_size; (void)ws_size;
  const int*   ctx   = (const int*)d_in[0];
  const int*   qry   = (const int*)d_in[1];
  const int*   cmask = (const int*)d_in[2];
  const int*   qmask = (const int*)d_in[3];
  const float* emb   = (const float*)d_in[4];
  const float* attw  = (const float*)d_in[23];
  const float* osw   = (const float*)d_in[24];
  const float* osb   = (const float*)d_in[25];
  const float* oew   = (const float*)d_in[26];
  const float* oeb   = (const float*)d_in[27];
  float* outp = (float*)d_out;
  char* ws = (char*)d_ws;

  // ---- workspace arena (peak ~272 MB, buffers reused across stages) ----
  size_t off = 0;
  auto A = [&](size_t n) { size_t o = off; off += (n + 255) & ~(size_t)255; return o; };
  size_t o_encWih[2], o_encWhh[2], o_m1Wih[2], o_m1Whh[2], o_m2Wih[2], o_m2Whh[2];
  for (int d = 0; d < 2; ++d) {
    o_encWih[d] = A((size_t)512 * 128 * 2);
    o_encWhh[d] = A((size_t)512 * 128 * 2);
    o_m1Wih[d]  = A((size_t)512 * 1024 * 2);
    o_m1Whh[d]  = A((size_t)512 * 128 * 2);
    o_m2Wih[d]  = A((size_t)512 * 256 * 2);
    o_m2Whh[d]  = A((size_t)512 * 128 * 2);
  }
  size_t o_cEmb = A((size_t)32768 * 128 * 2);       // 8 MB
  size_t o_qEmb = A((size_t)4096 * 128 * 2);        // 1 MB
  size_t o_Xh   = A((size_t)32768 * 1024 * 2);      // 64 MB (g f16 / m1 f16)
  size_t o_xW0  = A((size_t)32768 * 512 * 4);       // 64 MB (dir0 xW / c2q scratch)
  size_t o_xW1  = A((size_t)32768 * 512 * 4);       // 64 MB (dir1 xW)
  size_t o_cenc = A((size_t)64 * 512 * 256 * 4);    // 32 MB (c_enc, later m2)
  size_t o_qenc = A((size_t)64 * 64 * 256 * 4);     // 4 MB
  size_t o_m    = A((size_t)64 * 512 * 256 * 4);    // 32 MB (m1 out)

  _Float16* Xh   = (_Float16*)(ws + o_Xh);
  float*    xW0  = (float*)(ws + o_xW0);
  float*    xW1  = (float*)(ws + o_xW1);
  float*    cenc = (float*)(ws + o_cenc);
  float*    qenc = (float*)(ws + o_qenc);
  float*    m1o  = (float*)(ws + o_m);

  // ---- convert weights to f16 (K-padded) ----
  for (int d = 0; d < 2; ++d) {
    k_pad16<<<(512 * 128 + 255) / 256, 256, 0, stream>>>(
        (const float*)d_in[5 + 3 * d], (_Float16*)(ws + o_encWih[d]), 512, 100, 128);
    k_pad16<<<(512 * 128 + 255) / 256, 256, 0, stream>>>(
        (const float*)d_in[6 + 3 * d], (_Float16*)(ws + o_encWhh[d]), 512, 128, 128);
    k_pad16<<<(512 * 1024 + 255) / 256, 256, 0, stream>>>(
        (const float*)d_in[11 + 3 * d], (_Float16*)(ws + o_m1Wih[d]), 512, 1024, 1024);
    k_pad16<<<(512 * 128 + 255) / 256, 256, 0, stream>>>(
        (const float*)d_in[12 + 3 * d], (_Float16*)(ws + o_m1Whh[d]), 512, 128, 128);
    k_pad16<<<(512 * 256 + 255) / 256, 256, 0, stream>>>(
        (const float*)d_in[17 + 3 * d], (_Float16*)(ws + o_m2Wih[d]), 512, 256, 256);
    k_pad16<<<(512 * 128 + 255) / 256, 256, 0, stream>>>(
        (const float*)d_in[18 + 3 * d], (_Float16*)(ws + o_m2Whh[d]), 512, 128, 128);
  }

  // ---- embeddings ----
  k_embed<<<(32768 * 128 + 255) / 256, 256, 0, stream>>>(
      emb, ctx, (_Float16*)(ws + o_cEmb), 32768, 100, 128);
  k_embed<<<(4096 * 128 + 255) / 256, 256, 0, stream>>>(
      emb, qry, (_Float16*)(ws + o_qEmb), 4096, 100, 128);

  const size_t lstm_lds = 64 * 128 * sizeof(_Float16)   // h
                        + 64 * 128 * 4                  // c
                        + 64 * 512 * 4                  // gates
                        + 512 * 128 * sizeof(_Float16); // Whh  -> 304 KB

  // ---- encoder BiLSTM: context (both dirs concurrent), then query ----
  k_gemm_f16<<<32768 / 16 * 8 / 8, 256, 0, stream>>>(
      (_Float16*)(ws + o_cEmb), (_Float16*)(ws + o_encWih[0]), (const float*)d_in[7], xW0, 32768, 128, 512);
  k_gemm_f16<<<32768 / 16 * 8 / 8, 256, 0, stream>>>(
      (_Float16*)(ws + o_cEmb), (_Float16*)(ws + o_encWih[1]), (const float*)d_in[10], xW1, 32768, 128, 512);
  k_lstm2<<<2, 1024, lstm_lds, stream>>>(xW0, xW1, (_Float16*)(ws + o_encWhh[0]),
                                         (_Float16*)(ws + o_encWhh[1]), cenc, 512);
  k_gemm_f16<<<4096 / 16 * 8 / 8, 256, 0, stream>>>(
      (_Float16*)(ws + o_qEmb), (_Float16*)(ws + o_encWih[0]), (const float*)d_in[7], xW0, 4096, 128, 512);
  k_gemm_f16<<<4096 / 16 * 8 / 8, 256, 0, stream>>>(
      (_Float16*)(ws + o_qEmb), (_Float16*)(ws + o_encWih[1]), (const float*)d_in[10], xW1, 4096, 128, 512);
  k_lstm2<<<2, 1024, lstm_lds, stream>>>(xW0, xW1, (_Float16*)(ws + o_encWhh[0]),
                                         (_Float16*)(ws + o_encWhh[1]), qenc, 64);

  // ---- BiDAF attention -> g (f16, into Xh) ; c2q scratch reuses xW0 region ----
  const size_t bidaf_lds = (size_t)(64 * 256 + 768 + 64 + 512 + 8 * 256 + 8 * 64 + 256 + 256) * 4;
  k_bidaf<<<64, 256, bidaf_lds, stream>>>(cenc, qenc, cmask, qmask, attw, xW0, Xh);

  // ---- mod1 BiLSTM (K=1024), directions concurrent ----
  k_gemm_f16<<<32768 / 16 * 8 / 8, 256, 0, stream>>>(
      Xh, (_Float16*)(ws + o_m1Wih[0]), (const float*)d_in[13], xW0, 32768, 1024, 512);
  k_gemm_f16<<<32768 / 16 * 8 / 8, 256, 0, stream>>>(
      Xh, (_Float16*)(ws + o_m1Wih[1]), (const float*)d_in[16], xW1, 32768, 1024, 512);
  k_lstm2<<<2, 1024, lstm_lds, stream>>>(xW0, xW1, (_Float16*)(ws + o_m1Whh[0]),
                                         (_Float16*)(ws + o_m1Whh[1]), m1o, 512);

  // ---- m1 -> f16 (reuse Xh), then mod2 BiLSTM (K=256), out into cenc region ----
  k_pad16<<<(32768 * 256 + 255) / 256, 256, 0, stream>>>(m1o, Xh, 32768, 256, 256);
  float* m2o = cenc; // c_enc no longer needed
  k_gemm_f16<<<32768 / 16 * 8 / 8, 256, 0, stream>>>(
      Xh, (_Float16*)(ws + o_m2Wih[0]), (const float*)d_in[19], xW0, 32768, 256, 512);
  k_gemm_f16<<<32768 / 16 * 8 / 8, 256, 0, stream>>>(
      Xh, (_Float16*)(ws + o_m2Wih[1]), (const float*)d_in[22], xW1, 32768, 256, 512);
  k_lstm2<<<2, 1024, lstm_lds, stream>>>(xW0, xW1, (_Float16*)(ws + o_m2Whh[0]),
                                         (_Float16*)(ws + o_m2Whh[1]), m2o, 512);

  // ---- output projections ----
  k_project<<<(64 * 512 + 255) / 256, 256, 0, stream>>>(m2o, osw, osb, oew, oeb, cmask, outp);
}
